// RMSNormSelfCrossAttention_5609227288853
// MI455X (gfx1250) — matrix operand order repaired and checked by reading, not compiled, not run on hardware
//
#include <hip/hip_runtime.h>

// ---------------------------------------------------------------------------
// RMSNorm self+cross linear attention for MI455X (gfx1250, wave32, WMMA).
//
// Shapes: B=4, L=2048, C=256, D=1024, H=16, DH=64, S=L+C=2304.
// ~78 GFLOP total, >97% in four GEMMs -> bf16 v_wmma_f32_16x16x32_bf16 with
// f32 accumulation. GEMM K-loop is double-buffered in LDS and uses the
// gfx1250 GLOBAL_LOAD_ASYNC_TO_LDS_B128 path (ASYNCcnt) when available:
// issue tile k+1 async, compute tile k, s_wait_asynccnt 0, one barrier/iter.
// ---------------------------------------------------------------------------

typedef unsigned short ushort_t;

typedef __bf16 v16bf __attribute__((ext_vector_type(16)));
typedef __bf16 v8bf  __attribute__((ext_vector_type(8)));
typedef float  v8f   __attribute__((ext_vector_type(8)));
typedef int    v4i   __attribute__((ext_vector_type(4)));

union BF16x16 { v16bf v; v8bf h[2]; };

#define RMS_EPS 1.1920928955078125e-07f
#define LIN_EPS 1e-06f

#ifndef __has_builtin
#define __has_builtin(x) 0
#endif
#if __has_builtin(__builtin_amdgcn_global_load_async_to_lds_b128) && \
    __has_builtin(__builtin_amdgcn_s_wait_asynccnt)
#define HAVE_ASYNC_LDS 1
#else
#define HAVE_ASYNC_LDS 0
#endif

#define AS1 __attribute__((address_space(1)))
#define AS3 __attribute__((address_space(3)))
typedef AS1 v4i* gv4i_p;   // global 128-bit pointer
typedef AS3 v4i* lv4i_p;   // LDS 128-bit pointer

__device__ __forceinline__ ushort_t f2bf(float f) {
  union { float f; unsigned u; } v; v.f = f;
  unsigned r = v.u + 0x7FFFu + ((v.u >> 16) & 1u);   // round-to-nearest-even
  return (ushort_t)(r >> 16);
}
__device__ __forceinline__ float bf2f(ushort_t u) {
  union { unsigned u; float f; } v; v.u = ((unsigned)u) << 16;
  return v.f;
}

__device__ __forceinline__ float blockSum(float v, float* sbuf) {
  #pragma unroll
  for (int o = 16; o; o >>= 1) v += __shfl_xor(v, o, 32);
  int lane = threadIdx.x & 31, w = threadIdx.x >> 5, nw = blockDim.x >> 5;
  __syncthreads();
  if (lane == 0) sbuf[w] = v;
  __syncthreads();
  float r = 0.f;
  for (int i = 0; i < nw; ++i) r += sbuf[i];
  return r;
}

// --------------------------- elementwise converts ---------------------------

__global__ __launch_bounds__(256) void k_f32_to_bf16(const float* __restrict__ in,
                                                     ushort_t* __restrict__ out, int n) {
  int i = blockIdx.x * 256 + threadIdx.x;
  if (i < n) out[i] = f2bf(in[i]);
}

// W (K x N, f32, row-major) -> Wt (N x K, bf16, row-major)
__global__ __launch_bounds__(256) void k_transpose_bf16(const float* __restrict__ in,
                                                        ushort_t* __restrict__ out,
                                                        int K, int N) {
  __shared__ float t[32][33];
  int k0 = blockIdx.y * 32, n0 = blockIdx.x * 32;
  int tx = threadIdx.x, ty = threadIdx.y;          // 32 x 8
  #pragma unroll
  for (int i = 0; i < 4; ++i) {
    int k = ty + i * 8;
    t[k][tx] = in[(size_t)(k0 + k) * N + n0 + tx];
  }
  __syncthreads();
  #pragma unroll
  for (int i = 0; i < 4; ++i) {
    int n = ty + i * 8;
    out[(size_t)(n0 + n) * K + k0 + tx] = f2bf(t[tx][n]);
  }
}

__global__ __launch_bounds__(256) void k_zero_f32(float* p, int n) {
  int i = blockIdx.x * 256 + threadIdx.x;
  if (i < n) p[i] = 0.f;
}

// --------------------------------- mask ------------------------------------
// k_mask[b][s] = attn_mask[b,0,s] for s<L ; = OR_j attn_mask[b,0,j] for s>=L.
__global__ __launch_bounds__(256) void k_build_mask(const unsigned char* __restrict__ am,
                                                    float* __restrict__ msk,
                                                    int L, int S) {
  int b = blockIdx.x;
  __shared__ int anyf;
  if (threadIdx.x == 0) anyf = 0;
  __syncthreads();
  int loc = 0;
  for (int s = threadIdx.x; s < L; s += blockDim.x) {
    unsigned char v = am[(size_t)b * L * L + s];
    msk[(size_t)b * S + s] = v ? 1.f : 0.f;
    loc |= v;
  }
  if (loc) atomicOr(&anyf, 1);
  __syncthreads();
  float a = anyf ? 1.f : 0.f;
  for (int s = L + threadIdx.x; s < S; s += blockDim.x)
    msk[(size_t)b * S + s] = a;
}

// ------------------------------ bf16 TN GEMM --------------------------------
// C[M x N] = A[M x K](bf16,row) * Bt[N x K](bf16,row)^T + bias.
// Block 256 thr = 8 waves; tile 128x128; wave tile 32(M) x 64(N); K-step 32.
// Double-buffered LDS; async global->LDS copies when available.
#define LDS_STRIDE 40                 // halves per row: conflict-free frag reads
#define LDS_TILE   (128 * LDS_STRIDE) // 5120 halves per buffer

__global__ __launch_bounds__(256) void k_gemm_bf16_tn(const ushort_t* __restrict__ A,
                                                      const ushort_t* __restrict__ Bt,
                                                      const float* __restrict__ bias,
                                                      float* __restrict__ C,
                                                      int M, int N, int K) {
  __shared__ ushort_t lA[2 * LDS_TILE];
  __shared__ ushort_t lB[2 * LDS_TILE];

  const int tid = threadIdx.x, lane = tid & 31, wv = tid >> 5;
  const int wm = wv & 3, wn = wv >> 2;
  const int m0 = blockIdx.y * 128, n0 = blockIdx.x * 128;

  v8f acc[2][4];
  #pragma unroll
  for (int i = 0; i < 2; ++i)
    #pragma unroll
    for (int j = 0; j < 4; ++j)
      #pragma unroll
      for (int r = 0; r < 8; ++r) acc[i][j][r] = 0.f;

  const int lr = tid >> 1;            // 0..127 (tile row)
  const int lc = (tid & 1) * 16;      // 0 or 16 halves
  const ushort_t* gA = A  + (size_t)(m0 + lr) * K + lc;
  const ushort_t* gB = Bt + (size_t)(n0 + lr) * K + lc;
  const int ldsoff = lr * LDS_STRIDE + lc;
  const int nk = K >> 5;

#if HAVE_ASYNC_LDS
  // -------- async pipeline: issue k+1, compute k, one barrier per iter ------
  {
    const int o0 = ldsoff;
    __builtin_amdgcn_global_load_async_to_lds_b128((gv4i_p)(gA),     (lv4i_p)&lA[o0],     0, 0);
    __builtin_amdgcn_global_load_async_to_lds_b128((gv4i_p)(gA + 8), (lv4i_p)&lA[o0 + 8], 0, 0);
    __builtin_amdgcn_global_load_async_to_lds_b128((gv4i_p)(gB),     (lv4i_p)&lB[o0],     0, 0);
    __builtin_amdgcn_global_load_async_to_lds_b128((gv4i_p)(gB + 8), (lv4i_p)&lB[o0 + 8], 0, 0);
    __builtin_amdgcn_s_wait_asynccnt(0);
    __syncthreads();
  }
#else
  // -------- register-staged prologue ---------------------------------------
  {
    const uint4* pa = reinterpret_cast<const uint4*>(gA);
    const uint4* pb = reinterpret_cast<const uint4*>(gB);
    uint4 a0 = pa[0], a1 = pa[1], b0 = pb[0], b1 = pb[1];
    *reinterpret_cast<uint4*>(&lA[ldsoff])     = a0;
    *reinterpret_cast<uint4*>(&lA[ldsoff + 8]) = a1;
    *reinterpret_cast<uint4*>(&lB[ldsoff])     = b0;
    *reinterpret_cast<uint4*>(&lB[ldsoff + 8]) = b1;
    __syncthreads();
  }
#endif

  for (int kt = 0; kt < nk; ++kt) {
    const int cur = kt & 1, nxt = cur ^ 1;
    const int cbase = cur * LDS_TILE;

#if HAVE_ASYNC_LDS
    if (kt + 1 < nk) {                       // async copy of tile k+1
      const ushort_t* na = gA + (size_t)(kt + 1) * 32;
      const ushort_t* nb = gB + (size_t)(kt + 1) * 32;
      const int no = nxt * LDS_TILE + ldsoff;
      __builtin_amdgcn_global_load_async_to_lds_b128((gv4i_p)(na),     (lv4i_p)&lA[no],     0, 0);
      __builtin_amdgcn_global_load_async_to_lds_b128((gv4i_p)(na + 8), (lv4i_p)&lA[no + 8], 0, 0);
      __builtin_amdgcn_global_load_async_to_lds_b128((gv4i_p)(nb),     (lv4i_p)&lB[no],     0, 0);
      __builtin_amdgcn_global_load_async_to_lds_b128((gv4i_p)(nb + 8), (lv4i_p)&lB[no + 8], 0, 0);
    }
#else
    uint4 a0, a1, b0, b1;
    if (kt + 1 < nk) {                       // overlap global loads with WMMA
      const uint4* pa = reinterpret_cast<const uint4*>(gA + (size_t)(kt + 1) * 32);
      const uint4* pb = reinterpret_cast<const uint4*>(gB + (size_t)(kt + 1) * 32);
      a0 = pa[0]; a1 = pa[1]; b0 = pb[0]; b1 = pb[1];
      if (kt + 2 < nk) {
        __builtin_prefetch(gA + (size_t)(kt + 2) * 32, 0, 1);
        __builtin_prefetch(gB + (size_t)(kt + 2) * 32, 0, 1);
      }
    }
#endif

    // A fragments (16x32): lane l -> row (l&15); halves [cb..cb+7],[cb+16..cb+23]
    BF16x16 af[2];
    const int cb = (lane >> 4) * 8;
    #pragma unroll
    for (int mi = 0; mi < 2; ++mi) {
      int r = wm * 32 + mi * 16 + (lane & 15);
      af[mi].h[0] = *reinterpret_cast<const v8bf*>(&lA[cbase + r * LDS_STRIDE + cb]);
      af[mi].h[1] = *reinterpret_cast<const v8bf*>(&lA[cbase + r * LDS_STRIDE + cb + 16]);
    }
    // B fragments (32x16): lane l -> col (l&15); halves [kb..kb+7],[kb+16..kb+23]
    BF16x16 bfr[4];
    const int kb = (lane >> 4) * 8;
    #pragma unroll
    for (int ni = 0; ni < 4; ++ni) {
      int n = wn * 64 + ni * 16 + (lane & 15);
      bfr[ni].h[0] = *reinterpret_cast<const v8bf*>(&lB[cbase + n * LDS_STRIDE + kb]);
      bfr[ni].h[1] = *reinterpret_cast<const v8bf*>(&lB[cbase + n * LDS_STRIDE + kb + 16]);
    }
    #pragma unroll
    for (int mi = 0; mi < 2; ++mi)
      #pragma unroll
      for (int ni = 0; ni < 4; ++ni)
        acc[mi][ni] = __builtin_amdgcn_wmma_f32_16x16x32_bf16(
            false, af[mi].v, false, bfr[ni].v, (short)0, acc[mi][ni], false, false);

#if HAVE_ASYNC_LDS
    if (kt + 1 < nk) {
      __builtin_amdgcn_s_wait_asynccnt(0);   // my copies landed
      __syncthreads();                       // everyone's copies landed
    }
#else
    if (kt + 1 < nk) {
      __syncthreads();                       // all waves done reading buf[nxt]
      const int no = nxt * LDS_TILE + ldsoff;
      *reinterpret_cast<uint4*>(&lA[no])     = a0;
      *reinterpret_cast<uint4*>(&lA[no + 8]) = a1;
      *reinterpret_cast<uint4*>(&lB[no])     = b0;
      *reinterpret_cast<uint4*>(&lB[no + 8]) = b1;
      __syncthreads();                       // stores visible for next iter
    }
#endif
  }

  // C/D layout: elem r of v8f -> row = r + 8*(lane>=16), col = lane&15
  #pragma unroll
  for (int mi = 0; mi < 2; ++mi)
    #pragma unroll
    for (int ni = 0; ni < 4; ++ni) {
      int rbase = m0 + wm * 32 + mi * 16 + ((lane >> 4) * 8);
      int col   = n0 + wn * 64 + ni * 16 + (lane & 15);
      float bv = bias ? bias[col] : 0.f;
      #pragma unroll
      for (int r = 0; r < 8; ++r)
        C[(size_t)(rbase + r) * N + col] = acc[mi][ni][r] + bv;
    }
}

// ----------------- RMSNorm + RoPE + ReLU feature maps (self) ----------------
__global__ __launch_bounds__(256) void k_norm_rope_self(const float* __restrict__ qkv,
                                                        const float* __restrict__ gq,
                                                        const float* __restrict__ gk,
                                                        const long long* __restrict__ start_pos,
                                                        const float* __restrict__ msk,
                                                        ushort_t* __restrict__ qf,
                                                        ushort_t* __restrict__ kf,
                                                        ushort_t* __restrict__ vf,
                                                        int L, int S) {
  __shared__ float red[8];
  const int m = blockIdx.x;
  const int b = m / L, l = m - b * L;
  const float* row = qkv + (size_t)m * 3072;

  float sq = 0.f, sk = 0.f;
  for (int d = threadIdx.x; d < 1024; d += 256) {
    float q = row[d];        sq += q * q;
    float k = row[1024 + d]; sk += k * k;
  }
  sq = blockSum(sq, red);
  sk = blockSum(sk, red);
  float invq = rsqrtf(sq * (1.f / 1024.f) + RMS_EPS);
  float invk = rsqrtf(sk * (1.f / 1024.f) + RMS_EPS);

  float pos  = (float)(start_pos[b] + (long long)l);
  float mval = msk[(size_t)b * S + l];

  for (int p = threadIdx.x; p < 512; p += 256) {
    int h = p >> 5, i = p & 31, d0 = 2 * p;      // d0 = h*64 + 2i
    // theta^{-2i/64} = 2^{-i * log2(1e4)/32}
    float fr = exp2f(-0.41524101186f * (float)i);
    float sn, cs;
    __sincosf(pos * fr, &sn, &cs);
    float q0 = row[d0] * invq * gq[d0];
    float q1 = row[d0 + 1] * invq * gq[d0 + 1];
    float k0 = row[1024 + d0] * invk * gk[d0];
    float k1 = row[1024 + d0 + 1] * invk * gk[d0 + 1];
    float qr0 = q0 * cs - q1 * sn, qr1 = q0 * sn + q1 * cs;
    float kr0 = k0 * cs - k1 * sn, kr1 = k0 * sn + k1 * cs;
    size_t base = (((size_t)(b * 16 + h) * S + l) * 64) + 2 * i;
    qf[base]     = f2bf(fmaxf(qr0, 0.f));
    qf[base + 1] = f2bf(fmaxf(qr1, 0.f));
    kf[base]     = f2bf(fmaxf(kr0, 0.f) * mval);
    kf[base + 1] = f2bf(fmaxf(kr1, 0.f) * mval);
  }
  for (int d = threadIdx.x; d < 1024; d += 256) {
    int h = d >> 6;
    vf[((size_t)(b * 16 + h) * S + l) * 64 + (d & 63)] = f2bf(row[2048 + d]);
  }
}

// -------------------- RMSNorm + ReLU feature maps (cross) -------------------
__global__ __launch_bounds__(256) void k_norm_ctx(const float* __restrict__ cqkv,
                                                  const float* __restrict__ gq,
                                                  const float* __restrict__ gk,
                                                  const float* __restrict__ msk,
                                                  ushort_t* __restrict__ qf,
                                                  ushort_t* __restrict__ kf,
                                                  ushort_t* __restrict__ vf,
                                                  int Cn, int L, int S) {
  __shared__ float red[8];
  const int m = blockIdx.x;
  const int b = m / Cn, c = m - b * Cn;
  const int s = L + c;
  const float* row = cqkv + (size_t)m * 3072;

  float sq = 0.f, sk = 0.f;
  for (int d = threadIdx.x; d < 1024; d += 256) {
    float q = row[d];        sq += q * q;
    float k = row[1024 + d]; sk += k * k;
  }
  sq = blockSum(sq, red);
  sk = blockSum(sk, red);
  float invq = rsqrtf(sq * (1.f / 1024.f) + RMS_EPS);
  float invk = rsqrtf(sk * (1.f / 1024.f) + RMS_EPS);
  float mval = msk[(size_t)b * S + L];   // any-mask for context keys

  for (int d = threadIdx.x; d < 1024; d += 256) {
    int h = d >> 6;
    size_t base = ((size_t)(b * 16 + h) * S + s) * 64 + (d & 63);
    qf[base] = f2bf(fmaxf(row[d] * invq * gq[d], 0.f));
    kf[base] = f2bf(fmaxf(row[1024 + d] * invk * gk[d], 0.f) * mval);
    vf[base] = f2bf(row[2048 + d]);
  }
}

// --------------- vk[e][d] = sum_s vpad[s][e] * kf[s][d]  (65x64) ------------
__global__ __launch_bounds__(256) void k_vk_partial(const ushort_t* __restrict__ kf,
                                                    const ushort_t* __restrict__ vf,
                                                    float* __restrict__ vk, int S) {
  __shared__ ushort_t kT[64 * 64];
  __shared__ ushort_t vT[64 * 64];
  const int bh = blockIdx.y, chunk = blockIdx.x;   // 6 chunks of 384 keys
  const int tid = threadIdx.x;
  const int d = tid & 63, e0 = tid >> 6;

  float accv[17];
  #pragma unroll
  for (int j = 0; j < 17; ++j) accv[j] = 0.f;

  const int lrr = tid >> 2, lcc = (tid & 3) * 16;
  for (int it = 0; it < 6; ++it) {
    int s0 = chunk * 384 + it * 64;
    size_t gro = ((size_t)bh * S + s0 + lrr) * 64 + lcc;
    __syncthreads();
    *reinterpret_cast<uint4*>(&kT[lrr * 64 + lcc])     = *reinterpret_cast<const uint4*>(kf + gro);
    *reinterpret_cast<uint4*>(&kT[lrr * 64 + lcc + 8]) = *reinterpret_cast<const uint4*>(kf + gro + 8);
    *reinterpret_cast<uint4*>(&vT[lrr * 64 + lcc])     = *reinterpret_cast<const uint4*>(vf + gro);
    *reinterpret_cast<uint4*>(&vT[lrr * 64 + lcc + 8]) = *reinterpret_cast<const uint4*>(vf + gro + 8);
    __syncthreads();
    for (int s = 0; s < 64; ++s) {
      float kv = bf2f(kT[s * 64 + d]);
      int j = 0;
      #pragma unroll
      for (int e = 0; e <= 64; e += 4, ++j) {
        int ee = e + e0;
        if (ee > 64) break;
        float vv = (ee == 64) ? 1.f : bf2f(vT[s * 64 + ee]);
        accv[j] += vv * kv;
      }
    }
  }
  int j = 0;
  for (int e = e0; e <= 64; e += 4, ++j)
    atomicAdd(&vk[(size_t)bh * 4160 + e * 64 + d], accv[j]);
}

// ------- res[s][e] = qf[s]·vk[e]; attn = res[0:64]/(res[64]+eps) ------------
__global__ __launch_bounds__(128) void k_attn_res(const ushort_t* __restrict__ qf,
                                                  const float* __restrict__ vk,
                                                  ushort_t* __restrict__ ax,
                                                  ushort_t* __restrict__ ac,
                                                  int S, int L, int Cn) {
  __shared__ float vkL[65 * 64];
  const int bh = blockIdx.y;
  const int s = blockIdx.x * 128 + threadIdx.x;
  for (int i = threadIdx.x; i < 4160; i += 128)
    vkL[i] = vk[(size_t)bh * 4160 + i];
  __syncthreads();

  float qv[64];
  const ushort_t* qrow = qf + ((size_t)bh * S + s) * 64;
  #pragma unroll
  for (int d = 0; d < 64; ++d) qv[d] = bf2f(qrow[d]);

  float den = 0.f;
  #pragma unroll
  for (int d = 0; d < 64; ++d) den += vkL[64 * 64 + d] * qv[d];
  float invden = 1.f / (den + LIN_EPS);

  const int b = bh >> 4, h = bh & 15;
  ushort_t* dst;
  if (s < L) dst = ax + (size_t)(b * L + s) * 1024 + h * 64;
  else       dst = ac + (size_t)(b * Cn + (s - L)) * 1024 + h * 64;

  for (int e = 0; e < 64; ++e) {
    float num = 0.f;
    #pragma unroll
    for (int d = 0; d < 64; ++d) num += vkL[e * 64 + d] * qv[d];
    dst[e] = f2bf(num * invden);
  }
}

// ------------------------------ final RMSNorm -------------------------------
__global__ __launch_bounds__(256) void k_rms_out(const float* __restrict__ in,
                                                 const float* __restrict__ g,
                                                 float* __restrict__ out) {
  __shared__ float red[8];
  const int m = blockIdx.x;
  const float* row = in + (size_t)m * 1024;
  float sq = 0.f;
  for (int d = threadIdx.x; d < 1024; d += 256) { float v = row[d]; sq += v * v; }
  sq = blockSum(sq, red);
  float inv = rsqrtf(sq * (1.f / 1024.f) + RMS_EPS);
  for (int d = threadIdx.x; d < 1024; d += 256)
    out[(size_t)m * 1024 + d] = row[d] * inv * g[d];
}

// ------------------------------- launcher -----------------------------------

extern "C" void kernel_launch(void* const* d_in, const int* in_sizes, int n_in,
                              void* d_out, int out_size, void* d_ws, size_t ws_size,
                              hipStream_t stream) {
  (void)in_sizes; (void)n_in; (void)out_size; (void)ws_size;
  constexpr int B = 4, L = 2048, Cn = 256, D = 1024, Hn = 16, S = L + Cn;
  constexpr int BL = B * L, BC = B * Cn;

  const float*         x      = (const float*)d_in[0];
  const float*         ctx    = (const float*)d_in[1];
  const unsigned char* am     = (const unsigned char*)d_in[2];
  const long long*     sp     = (const long long*)d_in[3];
  const float*         Wqkv   = (const float*)d_in[4];
  const float*         bqkv   = (const float*)d_in[5];
  const float*         g_q    = (const float*)d_in[6];
  const float*         g_k    = (const float*)d_in[7];
  const float*         cWqkv  = (const float*)d_in[8];
  const float*         cbqkv  = (const float*)d_in[9];
  const float*         cg_q   = (const float*)d_in[10];
  const float*         cg_k   = (const float*)d_in[11];
  const float*         Wout   = (const float*)d_in[12];
  const float*         bout   = (const float*)d_in[13];
  const float*         g_out  = (const float*)d_in[14];
  const float*         cWout  = (const float*)d_in[15];
  const float*         cbout  = (const float*)d_in[16];
  const float*         cg_out = (const float*)d_in[17];
  float* outp = (float*)d_out;

  char* w = (char*)d_ws;
  size_t off = 0;
  auto alloc = [&](size_t bytes) -> void* {
    void* p = w + off;
    off += (bytes + 255) & ~(size_t)255;
    return p;
  };

  ushort_t* xbf     = (ushort_t*)alloc((size_t)BL * D * 2);
  ushort_t* cbf     = (ushort_t*)alloc((size_t)BC * D * 2);
  ushort_t* WqkvT   = (ushort_t*)alloc((size_t)3 * D * D * 2);
  ushort_t* cWqkvT  = (ushort_t*)alloc((size_t)3 * D * D * 2);
  ushort_t* WoutT   = (ushort_t*)alloc((size_t)D * D * 2);
  ushort_t* cWoutT  = (ushort_t*)alloc((size_t)D * D * 2);
  float*    qkv     = (float*)alloc((size_t)BL * 3 * D * 4);
  float*    cqkv    = (float*)alloc((size_t)BC * 3 * D * 4);
  ushort_t* qf      = (ushort_t*)alloc((size_t)B * Hn * S * 64 * 2);
  ushort_t* kf      = (ushort_t*)alloc((size_t)B * Hn * S * 64 * 2);
  ushort_t* vf      = (ushort_t*)alloc((size_t)B * Hn * S * 64 * 2);
  float*    vkbuf   = (float*)alloc((size_t)B * Hn * 4160 * 4);
  ushort_t* attnxbf = (ushort_t*)alloc((size_t)BL * D * 2);
  ushort_t* attncbf = (ushort_t*)alloc((size_t)BC * D * 2);
  float*    outx    = (float*)alloc((size_t)BL * D * 4);
  float*    outc    = (float*)alloc((size_t)BC * D * 4);
  float*    msk     = (float*)alloc((size_t)B * S * 4);

  // 1. precision staging
  k_f32_to_bf16<<<(BL * D + 255) / 256, 256, 0, stream>>>(x, xbf, BL * D);
  k_f32_to_bf16<<<(BC * D + 255) / 256, 256, 0, stream>>>(ctx, cbf, BC * D);
  k_transpose_bf16<<<dim3(3 * D / 32, D / 32), dim3(32, 8), 0, stream>>>(Wqkv, WqkvT, D, 3 * D);
  k_transpose_bf16<<<dim3(3 * D / 32, D / 32), dim3(32, 8), 0, stream>>>(cWqkv, cWqkvT, D, 3 * D);
  k_transpose_bf16<<<dim3(D / 32, D / 32), dim3(32, 8), 0, stream>>>(Wout, WoutT, D, D);
  k_transpose_bf16<<<dim3(D / 32, D / 32), dim3(32, 8), 0, stream>>>(cWout, cWoutT, D, D);
  k_build_mask<<<B, 256, 0, stream>>>(am, msk, L, S);

  // 2. QKV projections (WMMA)
  k_gemm_bf16_tn<<<dim3(3 * D / 128, BL / 128), 256, 0, stream>>>(xbf, WqkvT, bqkv, qkv, BL, 3 * D, D);
  k_gemm_bf16_tn<<<dim3(3 * D / 128, BC / 128), 256, 0, stream>>>(cbf, cWqkvT, cbqkv, cqkv, BC, 3 * D, D);

  // 3. feature maps
  k_norm_rope_self<<<BL, 256, 0, stream>>>(qkv, g_q, g_k, sp, msk, qf, kf, vf, L, S);
  k_norm_ctx<<<BC, 256, 0, stream>>>(cqkv, cg_q, cg_k, msk, qf, kf, vf, Cn, L, S);

  // 4. linear attention state + readout
  k_zero_f32<<<(B * Hn * 4160 + 255) / 256, 256, 0, stream>>>(vkbuf, B * Hn * 4160);
  k_vk_partial<<<dim3(6, B * Hn), 256, 0, stream>>>(kf, vf, vkbuf, S);
  k_attn_res<<<dim3(S / 128, B * Hn), 128, 0, stream>>>(qf, vkbuf, attnxbf, attncbf, S, L, Cn);

  // 5. output projections (WMMA) + final norms
  k_gemm_bf16_tn<<<dim3(D / 128, BL / 128), 256, 0, stream>>>(attnxbf, WoutT, bout, outx, BL, D, D);
  k_gemm_bf16_tn<<<dim3(D / 128, BC / 128), 256, 0, stream>>>(attncbf, cWoutT, cbout, outc, BC, D, D);
  k_rms_out<<<BL, 256, 0, stream>>>(outx, g_out, outp);
  k_rms_out<<<BC, 256, 0, stream>>>(outc, cg_out, outp + (size_t)BL * D);
}
